// MultiHeadAttention_29600914604669
// MI455X (gfx1250) — compile-verified, matrix operand order checked by reading, not run on hardware
//
#include <hip/hip_runtime.h>
#include <stdint.h>

// Problem constants (from reference): B=2, S=2048, D=2048, H=16, HD=128
#define B_  2
#define S_  2048
#define D_  2048
#define H_  16
#define HD_ 128

typedef __bf16          bf16x16 __attribute__((ext_vector_type(16)));
typedef float           f32x8   __attribute__((ext_vector_type(8)));
typedef unsigned short  u16x16  __attribute__((ext_vector_type(16)));
typedef unsigned short  u16x8   __attribute__((ext_vector_type(8)));
typedef unsigned short  u16x4   __attribute__((ext_vector_type(4)));

// ---- optional CDNA5 async global->LDS path (guarded; falls back to reg staging) ----
#if defined(__has_builtin)
#if __has_builtin(__builtin_amdgcn_global_load_async_to_lds_b128) && \
    __has_builtin(__builtin_amdgcn_s_wait_asynccnt)
#define USE_ASYNC_LDS 1
#endif
#endif
#ifndef USE_ASYNC_LDS
#define USE_ASYNC_LDS 0
#endif

#if USE_ASYNC_LDS
// Exact parameter types per hipcc diagnostic:
//   param 1: '__attribute__((__vector_size__(4 * sizeof(int)))) int __device__ *' (AS1)
typedef int gv4i __attribute__((__vector_size__(4 * sizeof(int))));
typedef __attribute__((address_space(1))) gv4i* as1_v4i_ptr;
typedef __attribute__((address_space(3))) gv4i* as3_v4i_ptr;
#endif

__device__ __forceinline__ unsigned short f2bf(float f) {
  unsigned int u = __float_as_uint(f);
  u += 0x7fffu + ((u >> 16) & 1u);     // round-to-nearest-even
  return (unsigned short)(u >> 16);
}

__device__ __forceinline__ bf16x16 ld16(const unsigned short* p) {
  u16x16 r = *(const u16x16*)p;
  return __builtin_bit_cast(bf16x16, r);
}

// copy 16 bytes global -> LDS (async if available)
__device__ __forceinline__ void cp16_g2s(unsigned short* lds, const unsigned short* g) {
#if USE_ASYNC_LDS
  __builtin_amdgcn_global_load_async_to_lds_b128(
      (as1_v4i_ptr)(void*)g, (as3_v4i_ptr)(void*)lds, 0, 0);
#else
  *(u16x8*)lds = *(const u16x8*)g;
#endif
}

__device__ __forceinline__ void wait_async() {
#if USE_ASYNC_LDS
  __builtin_amdgcn_s_wait_asynccnt(0);
#endif
}

__device__ __forceinline__ f32x8 wmma_bf16(bf16x16 a, bf16x16 b, f32x8 c) {
  // D = A(16x32) * B(32x16) + C, f32 accumulate
  return __builtin_amdgcn_wmma_f32_16x16x32_bf16(false, a, false, b, (short)0, c,
                                                 false, false);
}

// ---------------------------------------------------------------------------
// Stage 0: f32 -> bf16 conversion
// ---------------------------------------------------------------------------
__global__ __launch_bounds__(256)
void cvt_f32_to_bf16_kernel(const float* __restrict__ in,
                            unsigned short* __restrict__ out, long n) {
  const long t = (long)blockIdx.x * blockDim.x + threadIdx.x;
  const long stride = (long)gridDim.x * blockDim.x;
  for (long i = t * 4; i < n; i += stride * 4) {
    const float4 f = *(const float4*)(in + i);
    u16x4 o;
    o[0] = f2bf(f.x); o[1] = f2bf(f.y); o[2] = f2bf(f.z); o[3] = f2bf(f.w);
    *(u16x4*)(out + i) = o;
  }
}

// ---------------------------------------------------------------------------
// Stage 1/4: GEMM  C[M,N] = A[M,K] (bf16, row-major) * W[N,K]^T (bf16, row-major)
// Block tile 128x256x32, 256 threads = 8 waves in 2x4 grid, wave tile 64x64.
// ---------------------------------------------------------------------------
template <bool OUT_F32>
__global__ __launch_bounds__(256)
void gemm_bt_kernel(const unsigned short* __restrict__ A,
                    const unsigned short* __restrict__ W,
                    float* __restrict__ Cf,
                    unsigned short* __restrict__ Cb,
                    int M, int N, int K) {
  constexpr int BM = 128, BN = 256, BK = 32;
  __shared__ alignas(32) unsigned short As[2][BM * BK];   // 8 KB x2
  __shared__ alignas(32) unsigned short Bs[2][BN * BK];   // 16 KB x2

  const int tid  = threadIdx.x;
  const int lane = tid & 31;
  const int wid  = tid >> 5;
  const int wr   = wid >> 2;   // 0..1
  const int wc   = wid & 3;    // 0..3
  const long bm  = (long)blockIdx.y * BM;
  const long bn  = (long)blockIdx.x * BN;

  const f32x8 vzero = {0.f, 0.f, 0.f, 0.f, 0.f, 0.f, 0.f, 0.f};
  f32x8 acc[4][4];
#pragma unroll
  for (int i = 0; i < 4; ++i)
#pragma unroll
    for (int j = 0; j < 4; ++j) acc[i][j] = vzero;

  const int nk = K / BK;

  auto stage = [&](int kt, int buf) {
    const long k0 = (long)kt * BK;
    // A tile: 128 rows x 32 cols = 512 16B chunks, 2 per thread
#pragma unroll
    for (int i = 0; i < 2; ++i) {
      const int c   = tid + i * 256;
      const int row = c >> 2;
      const int col = (c & 3) * 8;
      cp16_g2s(&As[buf][row * BK + col], &A[(bm + row) * K + k0 + col]);
    }
    // B tile: 256 rows x 32 cols = 1024 16B chunks, 4 per thread
#pragma unroll
    for (int i = 0; i < 4; ++i) {
      const int c   = tid + i * 256;
      const int row = c >> 2;
      const int col = (c & 3) * 8;
      cp16_g2s(&Bs[buf][row * BK + col], &W[(bn + row) * K + k0 + col]);
    }
  };

  stage(0, 0);
  wait_async();
  __syncthreads();

  for (int kt = 0; kt < nk; ++kt) {
    const int buf = kt & 1;
    if (kt + 1 < nk) stage(kt + 1, buf ^ 1);

    bf16x16 af[4], bfm[4];
#pragma unroll
    for (int i = 0; i < 4; ++i)
      af[i] = ld16(&As[buf][(wr * 64 + i * 16 + (lane & 15)) * BK + ((lane >> 4) << 4)]);
#pragma unroll
    for (int j = 0; j < 4; ++j)
      bfm[j] = ld16(&Bs[buf][(wc * 64 + j * 16 + (lane & 15)) * BK + ((lane >> 4) << 4)]);

#pragma unroll
    for (int i = 0; i < 4; ++i)
#pragma unroll
      for (int j = 0; j < 4; ++j)
        acc[i][j] = wmma_bf16(af[i], bfm[j], acc[i][j]);

    wait_async();
    __syncthreads();
  }

  // Epilogue. C layout: VGPR r -> row (r + 8*(lane>=16)), col lane%16.
#pragma unroll
  for (int i = 0; i < 4; ++i)
#pragma unroll
    for (int j = 0; j < 4; ++j)
#pragma unroll
      for (int r = 0; r < 8; ++r) {
        const long row = bm + wr * 64 + i * 16 + r + ((lane >> 4) << 3);
        const long col = bn + wc * 64 + j * 16 + (lane & 15);
        const float v = acc[i][j][r];
        if constexpr (OUT_F32) Cf[row * (long)N + col] = v;
        else                   Cb[row * (long)N + col] = f2bf(v);
      }
}

// ---------------------------------------------------------------------------
// Stage 3: flash attention. Grid (S/64, H, B), 128 threads = 4 waves.
// Wave w owns 16 query rows; d=128 kept in registers; 64-key tiles.
// ---------------------------------------------------------------------------
__global__ __launch_bounds__(128)
void flash_attn_kernel(const unsigned short* __restrict__ xq,
                       const unsigned short* __restrict__ xk,
                       const unsigned short* __restrict__ xv,
                       unsigned short* __restrict__ ctx) {
  __shared__ alignas(32) unsigned short Ks[64 * 128];   // [key][d]  16 KB
  __shared__ alignas(32) unsigned short Vt[128 * 64];   // [d][key]  16 KB (transposed)
  __shared__ alignas(32) unsigned short Ps[4][16 * 64]; // per-wave P staging, 8 KB

  const int tid  = threadIdx.x;
  const int lane = tid & 31;
  const int w    = tid >> 5;
  const int b    = blockIdx.z;
  const int h    = blockIdx.y;
  const int qb0  = blockIdx.x * 64;
  const float rscale = 0.08838834764831845f;  // 1/sqrt(128)

  // Q fragments (A-layout): lane holds row lane%16, 16 contiguous d per chunk
  const long qrow0 = (long)b * S_ + qb0 + w * 16;
  bf16x16 aq[4];
#pragma unroll
  for (int c = 0; c < 4; ++c)
    aq[c] = ld16(&xq[(qrow0 + (lane & 15)) * D_ + h * HD_ + c * 32 + ((lane >> 4) << 4)]);

  const f32x8 vzero = {0.f, 0.f, 0.f, 0.f, 0.f, 0.f, 0.f, 0.f};
  f32x8 oacc[8];
#pragma unroll
  for (int j = 0; j < 8; ++j) oacc[j] = vzero;
  float m8[8], l8[8];
#pragma unroll
  for (int r = 0; r < 8; ++r) { m8[r] = -3.0e38f; l8[r] = 0.f; }

  const int nkt = blockIdx.x + 1;  // causal: key tiles 0..diag
  for (int kt = 0; kt < nkt; ++kt) {
    const int  kbase = kt * 64;
    const long krow0 = (long)b * S_ + kbase;

    // Cooperative K (async->LDS) and V (transposed into LDS) tile load.
#pragma unroll
    for (int i = 0; i < 8; ++i) {
      const int c   = tid + i * 128;   // 1024 16B chunks
      const int row = c >> 4;          // 0..63
      const int col = (c & 15) * 8;    // 0..120
      cp16_g2s(&Ks[row * 128 + col], &xk[(krow0 + row) * D_ + h * HD_ + col]);
      const u16x8 vv = *(const u16x8*)&xv[(krow0 + row) * D_ + h * HD_ + col];
#pragma unroll
      for (int e = 0; e < 8; ++e) Vt[(col + e) * 64 + row] = vv[e];
    }
    wait_async();
    __syncthreads();

    // S = Q * K^T  (16 queries x 64 keys per wave)
    f32x8 sacc[4];
#pragma unroll
    for (int n = 0; n < 4; ++n) sacc[n] = vzero;
#pragma unroll
    for (int c = 0; c < 4; ++c)
#pragma unroll
      for (int n = 0; n < 4; ++n) {
        const bf16x16 bk =
            ld16(&Ks[(n * 16 + (lane & 15)) * 128 + c * 32 + ((lane >> 4) << 4)]);
        sacc[n] = wmma_bf16(aq[c], bk, sacc[n]);
      }

    // Online softmax (rows live across 16-lane halves in C layout)
    const bool diag = (kt == nkt - 1);
#pragma unroll
    for (int r = 0; r < 8; ++r) {
      const int rowl = r + ((lane >> 4) << 3);
      const int grow = qb0 + w * 16 + rowl;
      float rmax = -3.0e38f;
#pragma unroll
      for (int n = 0; n < 4; ++n) {
        float s = sacc[n][r] * rscale;
        if (diag) {
          const int gcol = kbase + n * 16 + (lane & 15);
          if (gcol > grow) s = -3.0e38f;
        }
        sacc[n][r] = s;
        rmax = fmaxf(rmax, s);
      }
#pragma unroll
      for (int off = 1; off < 16; off <<= 1)
        rmax = fmaxf(rmax, __shfl_xor(rmax, off, 32));
      const float mnew  = fmaxf(m8[r], rmax);
      const float alpha = __expf(m8[r] - mnew);
      m8[r] = mnew;
      float rsum = 0.f;
#pragma unroll
      for (int n = 0; n < 4; ++n) {
        const float p = __expf(sacc[n][r] - mnew);
        sacc[n][r] = p;
        rsum += p;
      }
#pragma unroll
      for (int off = 1; off < 16; off <<= 1) rsum += __shfl_xor(rsum, off, 32);
      l8[r] = l8[r] * alpha + rsum;
#pragma unroll
      for (int j = 0; j < 8; ++j) oacc[j][r] = oacc[j][r] * alpha;
    }

    // P: C-layout -> bf16 A-layout via LDS
#pragma unroll
    for (int n = 0; n < 4; ++n)
#pragma unroll
      for (int r = 0; r < 8; ++r) {
        const int rowl = r + ((lane >> 4) << 3);
        Ps[w][rowl * 64 + n * 16 + (lane & 15)] = f2bf(sacc[n][r]);
      }
    __syncthreads();

    // O += P * V
#pragma unroll
    for (int kc = 0; kc < 2; ++kc) {
      const bf16x16 ap =
          ld16(&Ps[w][(lane & 15) * 64 + kc * 32 + ((lane >> 4) << 4)]);
#pragma unroll
      for (int j = 0; j < 8; ++j) {
        const bf16x16 bv =
            ld16(&Vt[(j * 16 + (lane & 15)) * 64 + kc * 32 + ((lane >> 4) << 4)]);
        oacc[j] = wmma_bf16(ap, bv, oacc[j]);
      }
    }
    __syncthreads();
  }

  // Normalize and write context (bf16), layout [B,S,H*d]
#pragma unroll
  for (int r = 0; r < 8; ++r) {
    const float inv  = 1.0f / l8[r];
    const int   rowl = r + ((lane >> 4) << 3);
    const long  grow = (long)b * S_ + qb0 + w * 16 + rowl;
#pragma unroll
    for (int j = 0; j < 8; ++j)
      ctx[grow * D_ + h * HD_ + j * 16 + (lane & 15)] = f2bf(oacc[j][r] * inv);
  }
}

// ---------------------------------------------------------------------------
// Launcher
// ---------------------------------------------------------------------------
extern "C" void kernel_launch(void* const* d_in, const int* in_sizes, int n_in,
                              void* d_out, int out_size, void* d_ws, size_t ws_size,
                              hipStream_t stream) {
  (void)in_sizes; (void)n_in; (void)out_size; (void)ws_size;

  const float* q  = (const float*)d_in[0];
  const float* k  = (const float*)d_in[1];
  const float* v  = (const float*)d_in[2];
  const float* wq = (const float*)d_in[3];
  const float* wk = (const float*)d_in[4];
  const float* wv = (const float*)d_in[5];
  const float* wo = (const float*)d_in[6];

  const long BS = (long)B_ * S_;  // 4096
  const long NX = BS * D_;        // 8M elems
  const long NW = (long)D_ * D_;  // 4M elems

  unsigned short* ws  = (unsigned short*)d_ws;
  unsigned short* qb  = ws;
  unsigned short* kb  = qb  + NX;
  unsigned short* vb  = kb  + NX;
  unsigned short* wqb = vb  + NX;
  unsigned short* wkb = wqb + NW;
  unsigned short* wvb = wkb + NW;
  unsigned short* wob = wvb + NW;
  unsigned short* xq  = wob + NW;
  unsigned short* xk  = xq  + NX;
  unsigned short* xv  = xk  + NX;
  unsigned short* ctx = xv  + NX;   // total ~151 MB of d_ws

  const dim3 cvtg(2048), cvtb(256);
  cvt_f32_to_bf16_kernel<<<cvtg, cvtb, 0, stream>>>(q,  qb,  NX);
  cvt_f32_to_bf16_kernel<<<cvtg, cvtb, 0, stream>>>(k,  kb,  NX);
  cvt_f32_to_bf16_kernel<<<cvtg, cvtb, 0, stream>>>(v,  vb,  NX);
  cvt_f32_to_bf16_kernel<<<cvtg, cvtb, 0, stream>>>(wq, wqb, NW);
  cvt_f32_to_bf16_kernel<<<cvtg, cvtb, 0, stream>>>(wk, wkb, NW);
  cvt_f32_to_bf16_kernel<<<cvtg, cvtb, 0, stream>>>(wv, wvb, NW);
  cvt_f32_to_bf16_kernel<<<cvtg, cvtb, 0, stream>>>(wo, wob, NW);

  const dim3 gb(256);
  const dim3 gg(D_ / 256, (unsigned)(BS / 128));   // (8, 32)
  gemm_bt_kernel<false><<<gg, gb, 0, stream>>>(qb, wqb, nullptr, xq, (int)BS, D_, D_);
  gemm_bt_kernel<false><<<gg, gb, 0, stream>>>(kb, wkb, nullptr, xk, (int)BS, D_, D_);
  gemm_bt_kernel<false><<<gg, gb, 0, stream>>>(vb, wvb, nullptr, xv, (int)BS, D_, D_);

  flash_attn_kernel<<<dim3(S_ / 64, H_, B_), dim3(128), 0, stream>>>(xq, xk, xv, ctx);

  gemm_bt_kernel<true><<<gg, gb, 0, stream>>>(ctx, wob, (float*)d_out, nullptr,
                                              (int)BS, D_, D_);
}
